// TfaSR_Point_39032662786456
// MI455X (gfx1250) — matrix operand order repaired and checked by reading, not compiled, not run on hardware
//
#include <hip/hip_runtime.h>
#include <hip/hip_bf16.h>
#include <stdint.h>

// MI455X (gfx1250) implementation.
// All conv3x3 / DCN GEMMs run through v_wmma_f32_16x16x32_bf16 implicit GEMM:
// fp32 activations in HBM, bf16 tiles staged in LDS, fp32 accumulation.
// Round-2: conv GEMM is a register-staged double-buffered pipeline (one
// barrier per K-chunk; global loads for chunk k+1 overlap WMMA on chunk k),
// and all A-fragments are hoisted before the WMMA burst so ds_loads clause
// together instead of serializing on s_wait_dscnt per WMMA.

#define HH 256
#define WW 256
#define BB 2
#define CC 64
#define HWSZ (HH*WW)
#define EPSV 1e-5f

typedef __bf16 bf16x16 __attribute__((ext_vector_type(16)));
typedef float f32x8    __attribute__((ext_vector_type(8)));

union FragB { unsigned u[8]; bf16x16 v; };
union FragC { float    f[8]; f32x8   v; };

__device__ __forceinline__ unsigned short f2bf(float f) {
  union { float f; unsigned u; } x; x.f = f;
  unsigned r = x.u + 0x7FFFu + ((x.u >> 16) & 1u);
  return (unsigned short)(r >> 16);
}

__device__ __forceinline__ float bilin_corner(const float* __restrict__ ch, int iy, int ix) {
  bool valid = (iy >= 0) && (iy < HH) && (ix >= 0) && (ix < WW);
  int cy = min(max(iy, 0), HH - 1);
  int cx = min(max(ix, 0), WW - 1);
  float v = ch[cy * WW + cx];
  return valid ? v : 0.f;
}

// ---------------------------------------------------------------------------
// Elementwise pre-processing: conv0 (1x1 over [Slope,Distance,Level]) + mask.
// ---------------------------------------------------------------------------
__global__ __launch_bounds__(256) void preproc_kernel(
    const float* __restrict__ LowDEM, const float* __restrict__ Point_Ele,
    const float* __restrict__ Slope,  const float* __restrict__ Distance,
    const float* __restrict__ Level,  const float* __restrict__ w0,
    const float* __restrict__ b0, float* __restrict__ x0) {
  int i = blockIdx.x * 256 + threadIdx.x;            // grid covers B*H*W exactly
  float err = Slope[i] * w0[0] + Distance[i] * w0[1] + Level[i] * w0[2] + b0[0];
  float lm  = (Level[i] != 0.f) ? 1.f : 0.f;
  x0[i] = LowDEM[i] * (1.f - lm) + Point_Ele[i] + err * lm;
}

// ---------------------------------------------------------------------------
// conv1: 9x9, Cin=1 -> Cout=64, pad 4. Direct conv (1.4 GFLOP, non-critical).
// ---------------------------------------------------------------------------
__global__ __launch_bounds__(256) void conv1_9x9_kernel(
    const float* __restrict__ x0, const float* __restrict__ w,
    const float* __restrict__ bias, float* __restrict__ out) {
  __shared__ float tile[24 * 24];
  __shared__ float wsh[64 * 81];
  int tid = threadIdx.x;
  int bx = blockIdx.x, by = blockIdx.y, b = blockIdx.z;
  for (int idx = tid; idx < 576; idx += 256) {
    int ry = idx / 24, rx = idx % 24;
    int yy = by * 16 + ry - 4, xx = bx * 16 + rx - 4;
    tile[idx] = (yy >= 0 && yy < HH && xx >= 0 && xx < WW)
                    ? x0[(size_t)b * HWSZ + yy * WW + xx] : 0.f;
  }
  for (int idx = tid; idx < 5184; idx += 256) wsh[idx] = w[idx];
  __syncthreads();
  int tx = tid & 15, ty = tid >> 4;
  int oy = by * 16 + ty, ox = bx * 16 + tx;
  for (int c = 0; c < 64; ++c) {
    float acc = bias[c];
    #pragma unroll 9
    for (int t = 0; t < 81; ++t)
      acc += tile[(ty + t / 9) * 24 + tx + t % 9] * wsh[c * 81 + t];
    out[(size_t)(b * 64 + c) * HWSZ + oy * WW + ox] = acc;
  }
}

// ---------------------------------------------------------------------------
// Weight repack: fp32 [cout][cin][3][3] -> bf16 [coutPad][tap*64+cin]
// (k = tap*64+cin so a 32-wide K chunk is contiguous bf16 pairs). Pads bias.
// ---------------------------------------------------------------------------
__global__ __launch_bounds__(256) void repack_kernel(
    const float* __restrict__ w, const float* __restrict__ bias,
    unsigned short* __restrict__ wpack, float* __restrict__ bpad,
    int coutReal, int coutPad) {
  int idx = blockIdx.x * 256 + threadIdx.x;
  int tot = coutPad * 576;
  if (idx < tot) {
    int co = idx / 576, k = idx % 576;
    int t = k >> 6, cin = k & 63;
    unsigned short v = 0;
    if (co < coutReal) v = f2bf(w[(size_t)co * 576 + cin * 9 + t]);
    wpack[(size_t)co * 576 + (t << 6) + cin] = v;
  }
  if (idx < coutPad) bpad[idx] = (idx < coutReal) ? bias[idx] : 0.f;
}

// ---------------------------------------------------------------------------
// Register staging of one K-chunk (global -> VGPRs) for the conv GEMM.
// ---------------------------------------------------------------------------
template <int NSUB>
__device__ __forceinline__ void conv_stage(
    const float* __restrict__ inb, const unsigned short* __restrict__ wpack,
    int kc, int tid, int h, int w0, float (&pv)[16],
    unsigned short (&wv)[2 * NSUB]) {
  int t = kc >> 1, cb = (kc & 1) << 5;
  int ky = t / 3 - 1, kx = t % 3 - 1;
  int yy = h + ky;
  #pragma unroll
  for (int j = 0; j < 16; ++j) {
    int flat = tid + (j << 8);
    int c = flat >> 7, p = flat & 127;
    int xx = w0 + p + kx;
    float v = 0.f;
    if (yy >= 0 && yy < HH && xx >= 0 && xx < WW)
      v = inb[(size_t)(cb + c) * HWSZ + yy * WW + xx];
    pv[j] = v;
  }
  #pragma unroll
  for (int j = 0; j < 2 * NSUB; ++j) {
    int idx = tid + (j << 8);
    wv[j] = wpack[(size_t)(idx >> 5) * 576 + (t << 6) + cb + (idx & 31)];
  }
}

// ---------------------------------------------------------------------------
// Implicit-GEMM 3x3 conv via WMMA (Cin=64, pad=1).
// D[cout][pixel] = sum_k W[cout][k] * patch[k][pixel], K = 9*64 = 576.
// Block = 256 threads (8 waves) -> tile of 128 pixels (one W-row segment) x
// NSUB*16 channels; wave w owns pixel-subtile w. Double-buffered LDS +
// register staging: one barrier per chunk, global loads of chunk k+1 overlap
// WMMA on chunk k (their wait lands at the next ds_store, under the XDL work).
// ---------------------------------------------------------------------------
template <int NSUB>
__global__ __launch_bounds__(256) void conv3_wmma_kernel(
    const float* __restrict__ in, const unsigned short* __restrict__ wpack,
    const float* __restrict__ bpad, float* __restrict__ out, int relu) {
  constexpr int OUTC = NSUB * 16;
  __shared__ __align__(16) unsigned short patch[2][128 * 36];
  __shared__ __align__(16) unsigned short wt[2][OUTC * 36];
  int tid = threadIdx.x;
  int bid = blockIdx.x;                 // 1024 blocks: b(1) | h(8) | wtile(1)
  int wtile = bid & 1, h = (bid >> 1) & (HH - 1), b = bid >> 9;
  int w0 = wtile * 128;
  int lane = tid & 31, wave = tid >> 5;

  FragC acc[NSUB];
  #pragma unroll
  for (int m = 0; m < NSUB; ++m)
    #pragma unroll
    for (int v = 0; v < 8; ++v) acc[m].f[v] = 0.f;

  const float* inb = in + (size_t)(b * CC) * HWSZ;

  float pv[16];
  unsigned short wv[2 * NSUB];
  conv_stage<NSUB>(inb, wpack, 0, tid, h, w0, pv, wv);

  for (int kc = 0; kc < 18; ++kc) {
    int buf = kc & 1;
    // ---- commit staged chunk kc to LDS (waits on its global loads here) ----
    #pragma unroll
    for (int j = 0; j < 16; ++j) {
      int flat = tid + (j << 8);
      patch[buf][(flat & 127) * 36 + (flat >> 7)] = f2bf(pv[j]);
    }
    #pragma unroll
    for (int j = 0; j < 2 * NSUB; ++j) {
      int idx = tid + (j << 8);
      wt[buf][(idx >> 5) * 36 + (idx & 31)] = wv[j];
    }
    // ---- issue global loads for chunk kc+1 (overlap with WMMA below) ----
    if (kc + 1 < 18) {
      conv_stage<NSUB>(inb, wpack, kc + 1, tid, h, w0, pv, wv);
      if (kc + 2 < 18) {
        int tn = (kc + 2) >> 1, cbn = ((kc + 2) & 1) << 5;
        __builtin_prefetch(wpack + (tn << 6) + cbn, 0, 0);  // global_prefetch_b8
      }
    }
    __syncthreads();

    // ---- WMMA on chunk kc: hoist all fragments, then burst the MACs ----
    int prow = (wave << 4) + (lane & 15);
    int kb2 = (lane >> 4) << 3;
    FragB bfr;
    #pragma unroll
    for (int v = 0; v < 8; ++v)
      bfr.u[v] = *(const unsigned*)&patch[buf][prow * 36 + kb2 + ((v < 4) ? (v << 1) : 16 + ((v - 4) << 1))];
    FragB af[NSUB];
    #pragma unroll
    for (int mi = 0; mi < NSUB; ++mi) {
      int crow = (mi << 4) + (lane & 15);
      int kh = (lane >> 4) << 3;
      #pragma unroll
      for (int v = 0; v < 8; ++v) {
        int kb = ((v < 4) ? (v << 1) : 16 + ((v - 4) << 1)) + kh;
        af[mi].u[v] = *(const unsigned*)&wt[buf][crow * 36 + kb];
      }
    }
    #pragma unroll
    for (int mi = 0; mi < NSUB; ++mi)
      acc[mi].v = __builtin_amdgcn_wmma_f32_16x16x32_bf16(
          false, af[mi].v, false, bfr.v, (short)0, acc[mi].v, false, false);
  }

  // ---- epilogue: bias (+relu), fp32 store, contiguous along W ----
  int px = (wave << 4) + (lane & 15);
  #pragma unroll
  for (int mi = 0; mi < NSUB; ++mi) {
    #pragma unroll
    for (int v = 0; v < 8; ++v) {
      int co = (mi << 4) + v + ((lane >> 4) << 3);
      float val = acc[mi].f[v] + bpad[co];
      if (relu) val = fmaxf(val, 0.f);
      out[(size_t)(b * OUTC + co) * HWSZ + h * WW + w0 + px] = val;
    }
  }
}

// ---------------------------------------------------------------------------
// DCN main GEMM: same WMMA pipeline, but the patch tile is built by bilinear
// sampling x at (h+ky+dy, w+kx+dx) scaled by sigmoid mask (from om).
// Gather-limited; keeps the simpler three-barrier structure, fragments hoisted.
// ---------------------------------------------------------------------------
template <int NSUB>
__global__ __launch_bounds__(256) void dcn_wmma_kernel(
    const float* __restrict__ in, const float* __restrict__ om,
    const unsigned short* __restrict__ wpack, const float* __restrict__ bpad,
    float* __restrict__ out, int relu) {
  constexpr int OUTC = NSUB * 16;
  __shared__ __align__(16) unsigned short patch[128 * 36];
  __shared__ __align__(16) unsigned short wt[OUTC * 36];
  __shared__ float sy[128], sx[128], sm[128];
  int tid = threadIdx.x;
  int bid = blockIdx.x;
  int wtile = bid & 1, h = (bid >> 1) & (HH - 1), b = bid >> 9;
  int w0 = wtile * 128;
  int lane = tid & 31, wave = tid >> 5;

  FragC acc[NSUB];
  #pragma unroll
  for (int m = 0; m < NSUB; ++m)
    #pragma unroll
    for (int v = 0; v < 8; ++v) acc[m].f[v] = 0.f;

  const float* inb = in + (size_t)(b * CC) * HWSZ;

  for (int kc = 0; kc < 18; ++kc) {
    int t = kc >> 1, cb = (kc & 1) << 5;
    int ky = t / 3 - 1, kx = t % 3 - 1;
    __syncthreads();
    if ((kc & 1) == 0 && tid < 128) {   // per-tap offsets/mask, both 32-ch halves
      int p = tid;
      size_t ob = (size_t)(b * 32) * HWSZ + (size_t)h * WW + w0 + p;
      float dy = om[ob + (size_t)t * HWSZ];
      float dx = om[ob + (size_t)(9 + t) * HWSZ];
      float mv = om[ob + (size_t)(18 + t) * HWSZ];
      sy[p] = (float)(h + ky) + dy;
      sx[p] = (float)(w0 + p + kx) + dx;
      sm[p] = 1.f / (1.f + __expf(-mv));
    }
    __syncthreads();
    #pragma unroll
    for (int j = 0; j < 16; ++j) {
      int flat = tid + (j << 8);
      int c = flat >> 7, p = flat & 127;
      const float* ch = inb + (size_t)(cb + c) * HWSZ;
      float ys = sy[p], xs = sx[p];
      float y0f = floorf(ys), x0f = floorf(xs);
      float wy = ys - y0f, wx = xs - x0f;
      int iy = (int)y0f, ix = (int)x0f;
      float v00 = bilin_corner(ch, iy,     ix);
      float v01 = bilin_corner(ch, iy,     ix + 1);
      float v10 = bilin_corner(ch, iy + 1, ix);
      float v11 = bilin_corner(ch, iy + 1, ix + 1);
      float s = (v00 * (1.f - wy) + v10 * wy) * (1.f - wx)
              + (v01 * (1.f - wy) + v11 * wy) * wx;
      patch[p * 36 + c] = f2bf(s * sm[p]);
    }
    #pragma unroll
    for (int j = 0; j < 2 * NSUB; ++j) {
      int idx = tid + (j << 8);
      wt[(idx >> 5) * 36 + (idx & 31)] =
          wpack[(size_t)(idx >> 5) * 576 + (t << 6) + cb + (idx & 31)];
    }
    __syncthreads();

    int prow = (wave << 4) + (lane & 15);
    int kb2 = (lane >> 4) << 3;
    FragB bfr;
    #pragma unroll
    for (int v = 0; v < 8; ++v)
      bfr.u[v] = *(const unsigned*)&patch[prow * 36 + kb2 + ((v < 4) ? (v << 1) : 16 + ((v - 4) << 1))];
    FragB af[NSUB];
    #pragma unroll
    for (int mi = 0; mi < NSUB; ++mi) {
      int crow = (mi << 4) + (lane & 15);
      int kh = (lane >> 4) << 3;
      #pragma unroll
      for (int v = 0; v < 8; ++v) {
        int kb = ((v < 4) ? (v << 1) : 16 + ((v - 4) << 1)) + kh;
        af[mi].u[v] = *(const unsigned*)&wt[crow * 36 + kb];
      }
    }
    #pragma unroll
    for (int mi = 0; mi < NSUB; ++mi)
      acc[mi].v = __builtin_amdgcn_wmma_f32_16x16x32_bf16(
          false, af[mi].v, false, bfr.v, (short)0, acc[mi].v, false, false);
  }

  int px = (wave << 4) + (lane & 15);
  #pragma unroll
  for (int mi = 0; mi < NSUB; ++mi) {
    #pragma unroll
    for (int v = 0; v < 8; ++v) {
      int co = (mi << 4) + v + ((lane >> 4) << 3);
      float val = acc[mi].f[v] + bpad[co];
      if (relu) val = fmaxf(val, 0.f);
      out[(size_t)(b * OUTC + co) * HWSZ + h * WW + w0 + px] = val;
    }
  }
}

// ---------------------------------------------------------------------------
// Batchnorm statistics, deterministic two-pass (no fp atomics).
// ---------------------------------------------------------------------------
__global__ __launch_bounds__(256) void stats1_kernel(const float* __restrict__ t,
                                                     float* __restrict__ part) {
  __shared__ float rs[256], rq[256];
  int c = blockIdx.x, s = blockIdx.y, tid = threadIdx.x;
  float sum = 0.f, sq = 0.f;
  const float* t0 = t + (size_t)c * HWSZ;
  const float* t1 = t + (size_t)(CC + c) * HWSZ;
  #pragma unroll
  for (int j = 0; j < 4; ++j) {
    int p = (s << 10) + tid + (j << 8);
    float a = t0[p], b2 = t1[p];
    sum += a + b2; sq += a * a + b2 * b2;
  }
  rs[tid] = sum; rq[tid] = sq;
  __syncthreads();
  for (int st = 128; st > 0; st >>= 1) {
    if (tid < st) { rs[tid] += rs[tid + st]; rq[tid] += rq[tid + st]; }
    __syncthreads();
  }
  if (tid == 0) { part[c * 64 + s] = rs[0]; part[4096 + c * 64 + s] = rq[0]; }
}

__global__ void stats2_kernel(const float* __restrict__ part, float* __restrict__ mv) {
  int c = threadIdx.x;
  if (c < CC) {
    float sum = 0.f, sq = 0.f;
    for (int s = 0; s < 64; ++s) { sum += part[c * 64 + s]; sq += part[4096 + c * 64 + s]; }
    float mean = sum / 131072.f;
    float var = sq / 131072.f - mean * mean;
    mv[c] = mean; mv[CC + c] = var;
  }
}

__global__ __launch_bounds__(256) void bn_apply_kernel(
    const float* __restrict__ t, const float* __restrict__ mv,
    const float* __restrict__ g, const float* __restrict__ be,
    const float* __restrict__ res, float* __restrict__ out, int relu) {
  size_t i = (size_t)blockIdx.x * 256 + threadIdx.x;
  int c = (int)((i >> 16) & 63);
  float mean = mv[c], var = mv[CC + c];
  float v = (t[i] - mean) * rsqrtf(var + EPSV) * g[c] + be[c];
  if (res) v += res[i];
  if (relu) v = fmaxf(v, 0.f);
  out[i] = v;
}

// Final 1x1 conv on dc4's channel 0 (stored in a 16-channel padded buffer).
__global__ __launch_bounds__(256) void final_kernel(
    const float* __restrict__ buf, const float* __restrict__ w4,
    const float* __restrict__ b4, float* __restrict__ out) {
  int i = blockIdx.x * 256 + threadIdx.x;
  int b = i >> 16, p = i & 65535;
  out[i] = buf[(size_t)(b * 16) * HWSZ + p] * w4[0] + b4[0];
}

// ---------------------------------------------------------------------------
extern "C" void kernel_launch(void* const* d_in, const int* in_sizes, int n_in,
                              void* d_out, int out_size, void* d_ws, size_t ws_size,
                              hipStream_t stream) {
  (void)in_sizes; (void)n_in; (void)out_size; (void)ws_size;
  const float* LowDEM   = (const float*)d_in[0];
  const float* PointEle = (const float*)d_in[1];
  const float* Slope    = (const float*)d_in[2];
  const float* Distance = (const float*)d_in[3];
  const float* Level    = (const float*)d_in[4];
  const float* conv0_w  = (const float*)d_in[5];
  const float* conv0_b  = (const float*)d_in[6];
  const float* conv1_w  = (const float*)d_in[7];
  const float* conv1_b  = (const float*)d_in[8];
  const float* rb_w1    = (const float*)d_in[9];
  const float* rb_b1    = (const float*)d_in[10];
  const float* rb_g1    = (const float*)d_in[11];
  const float* rb_be1   = (const float*)d_in[12];
  const float* rb_w2    = (const float*)d_in[13];
  const float* rb_b2    = (const float*)d_in[14];
  const float* rb_g2    = (const float*)d_in[15];
  const float* rb_be2   = (const float*)d_in[16];
  const float* conv2_w  = (const float*)d_in[17];
  const float* conv2_b  = (const float*)d_in[18];
  const float* bn2_g    = (const float*)d_in[19];
  const float* bn2_be   = (const float*)d_in[20];
  const float* dc2_w  = (const float*)d_in[21];
  const float* dc2_b  = (const float*)d_in[22];
  const float* dc2_ow = (const float*)d_in[23];
  const float* dc2_ob = (const float*)d_in[24];
  const float* dc3_w  = (const float*)d_in[25];
  const float* dc3_b  = (const float*)d_in[26];
  const float* dc3_ow = (const float*)d_in[27];
  const float* dc3_ob = (const float*)d_in[28];
  const float* dc4_w  = (const float*)d_in[29];
  const float* dc4_b  = (const float*)d_in[30];
  const float* dc4_ow = (const float*)d_in[31];
  const float* dc4_ob = (const float*)d_in[32];
  const float* conv4_w = (const float*)d_in[33];
  const float* conv4_b = (const float*)d_in[34];
  float* outp = (float*)d_out;

  char* ws = (char*)d_ws;
  const size_t ACT = (size_t)BB * CC * HWSZ * sizeof(float);     // 33.55 MB
  const size_t OMS = (size_t)BB * 32 * HWSZ * sizeof(float);     // 16.78 MB
  float* A0 = (float*)(ws);
  float* A1 = (float*)(ws + ACT);
  float* A2 = (float*)(ws + 2 * ACT);
  float* A3 = (float*)(ws + 3 * ACT);
  float* OM = (float*)(ws + 4 * ACT);
  unsigned short* WPK = (unsigned short*)(ws + 4 * ACT + OMS);
  float* BPD  = (float*)(ws + 4 * ACT + OMS + 73728);
  float* PART = (float*)(ws + 4 * ACT + OMS + 73728 + 256);
  float* MV   = PART + 8192;
  float* X0   = OM;   // OM region unused until the DCN stage

  auto conv3_64 = [&](const float* in, const float* w, const float* bias,
                      float* out, int relu) {
    repack_kernel<<<144, 256, 0, stream>>>(w, bias, WPK, BPD, 64, 64);
    conv3_wmma_kernel<4><<<1024, 256, 0, stream>>>(in, WPK, BPD, out, relu);
  };
  auto run_stats = [&](const float* t) {
    stats1_kernel<<<dim3(64, 64), 256, 0, stream>>>(t, PART);
    stats2_kernel<<<1, 64, 0, stream>>>(PART, MV);
  };

  // stem
  preproc_kernel<<<512, 256, 0, stream>>>(LowDEM, PointEle, Slope, Distance,
                                          Level, conv0_w, conv0_b, X0);
  conv1_9x9_kernel<<<dim3(16, 16, 2), 256, 0, stream>>>(X0, conv1_w, conv1_b, A0);
  hipMemcpyAsync(A1, A0, ACT, hipMemcpyDeviceToDevice, stream);   // y = x

  // residual blocks (y lives in A1, x in A0)
  for (int i = 0; i < 16; ++i) {
    conv3_64(A1, rb_w1 + (size_t)i * CC * CC * 9, rb_b1 + i * CC, A2, 0);
    run_stats(A2);
    bn_apply_kernel<<<32768, 256, 0, stream>>>(A2, MV, rb_g1 + i * CC,
                                               rb_be1 + i * CC, nullptr, A3, 1);
    conv3_64(A3, rb_w2 + (size_t)i * CC * CC * 9, rb_b2 + i * CC, A2, 0);
    run_stats(A2);
    bn_apply_kernel<<<32768, 256, 0, stream>>>(A2, MV, rb_g2 + i * CC,
                                               rb_be2 + i * CC, A1, A1, 0);
  }

  // x = bn(conv2(y)) + x   -> A3
  conv3_64(A1, conv2_w, conv2_b, A2, 0);
  run_stats(A2);
  bn_apply_kernel<<<32768, 256, 0, stream>>>(A2, MV, bn2_g, bn2_be, A0, A3, 0);

  // DCN: offset conv (27->pad 32 channels), then sampled GEMM.
  auto dcn_om = [&](const float* xin, const float* owt, const float* ob) {
    repack_kernel<<<72, 256, 0, stream>>>(owt, ob, WPK, BPD, 27, 32);
    conv3_wmma_kernel<2><<<1024, 256, 0, stream>>>(xin, WPK, BPD, OM, 0);
  };

  // dc2: x = relu(dcn(x))   A3 -> A2
  dcn_om(A3, dc2_ow, dc2_ob);
  repack_kernel<<<144, 256, 0, stream>>>(dc2_w, dc2_b, WPK, BPD, 64, 64);
  dcn_wmma_kernel<4><<<1024, 256, 0, stream>>>(A3, OM, WPK, BPD, A2, 1);

  // dc3: A2 -> A3
  dcn_om(A2, dc3_ow, dc3_ob);
  repack_kernel<<<144, 256, 0, stream>>>(dc3_w, dc3_b, WPK, BPD, 64, 64);
  dcn_wmma_kernel<4><<<1024, 256, 0, stream>>>(A2, OM, WPK, BPD, A3, 1);

  // dc4: 1 real out channel, padded to 16.  A3 -> A2
  dcn_om(A3, dc4_ow, dc4_ob);
  repack_kernel<<<36, 256, 0, stream>>>(dc4_w, dc4_b, WPK, BPD, 1, 16);
  dcn_wmma_kernel<1><<<1024, 256, 0, stream>>>(A3, OM, WPK, BPD, A2, 0);

  final_kernel<<<512, 256, 0, stream>>>(A2, conv4_w, conv4_b, outp);
}
